// LPKTNet_48979807044204
// MI455X (gfx1250) — matrix operand order repaired and checked by reading, not compiled, not run on hardware
//
#include <hip/hip_runtime.h>
#include <hip/hip_bf16.h>

typedef __attribute__((ext_vector_type(16))) _Float16 v16h;
typedef __attribute__((ext_vector_type(4)))  _Float16 v4h;
typedef __attribute__((ext_vector_type(8)))  float    v8f;

#define S_LEN 64
#define NQ1   513
#define NQ_PAD 528

__device__ __forceinline__ float sigf(float x) { return 1.0f / (1.0f + __expf(-x)); }

union FragU { v16h v; uint4 u[2]; };

// Pack 4 f32 -> 4 f16 and store as one 8-byte LDS write (dst must be 8B aligned).
__device__ __forceinline__ void store_h4(_Float16* dst, float4 v) {
  v4h h;
  h.x = (_Float16)v.x; h.y = (_Float16)v.y;
  h.z = (_Float16)v.z; h.w = (_Float16)v.w;
  *(v4h*)dst = h;
}

// A-fragment (16x32 f16) from row-major LDS tile, per ISA 7.12.2 layout:
// lane%16 = M row; halves 0..7 -> K = k0 + 8*(lane/16) + i ; halves 8..15 -> +16
__device__ __forceinline__ v16h load_a_lds(const _Float16* lds, int ldk, int k0) {
  int lane = threadIdx.x & 31;
  int m = lane & 15, hi = lane >> 4;
  const _Float16* p = lds + m * ldk + k0 + 8 * hi;
  FragU f;
  f.u[0] = *(const uint4*)(p);
  f.u[1] = *(const uint4*)(p + 16);
  return f.v;
}

// B-fragment (32x16 f16) from a pre-swizzled weight buffer: each lane's 16
// halves are stored contiguously -> two 16B global loads.
__device__ __forceinline__ v16h load_b_swz(const _Float16* wswz, int kt, int nt, int nNT) {
  int lane = threadIdx.x & 31;
  const _Float16* p = wswz + (((size_t)(kt * nNT + nt) * 32 + lane) << 4);
  FragU f;
  f.u[0] = *(const uint4*)(p);
  f.u[1] = *(const uint4*)(p + 8);
  return f.v;
}

__device__ __forceinline__ v8f wmma_f16(v16h a, v16h b, v8f c) {
  return __builtin_amdgcn_wmma_f32_16x16x32_f16(false, a, false, b, (short)0, c, false, false);
}

// ---------------- init / prep kernels ----------------

__global__ void k_zero(float* p, long long n) {
  long long i = (long long)blockIdx.x * blockDim.x + threadIdx.x;
  long long stride = (long long)gridDim.x * blockDim.x;
  for (; i < n; i += stride) p[i] = 0.0f;
}

// Pre-swizzle a (K x 128) f32 weight into B-fragment order f16.
// dst index j = ((kt*8 + nt)*32 + lane)*16 + i ;
// source k = kt*32 + 16*(i/8) + 8*(lane/16) + i%8, n = nt*16 + lane%16.
__global__ void k_swz(const float* __restrict__ W, _Float16* __restrict__ dst, int nKT) {
  int total = nKT * 8 * 512;
  for (int j = blockIdx.x * blockDim.x + threadIdx.x; j < total;
       j += gridDim.x * blockDim.x) {
    int i = j & 15, lane = (j >> 4) & 31, nt = (j >> 9) & 7, kt = j >> 12;
    int k = kt * 32 + ((i >> 3) << 4) + ((lane >> 4) << 3) + (i & 7);
    int n = (nt << 4) + (lane & 15);
    dst[j] = (_Float16)W[k * 128 + n];
  }
}

// Collapse W5 over its output dim: w5r[k] = sum_j W5[k][j]; w5r[256] = sum_j b5[j]
__global__ void k_w5row(const float* __restrict__ W5, const float* __restrict__ b5,
                        float* __restrict__ w5r) {
  int k = blockIdx.x * blockDim.x + threadIdx.x;
  if (k < 256) {
    float s = 0.f;
    for (int j = 0; j < 128; ++j) s += W5[k * 128 + j];
    w5r[k] = s;
  } else if (k == 256) {
    float s = 0.f;
    for (int j = 0; j < 128; ++j) s += b5[j];
    w5r[256] = s;
  }
}

// ---------------- K1: learning_all = [e_emb|at_emb|a_rep] @ W1 + b1 ----------------
// grid.x = 512 M-tiles of 16 rows; 256 threads = 8 waves, wave w owns N-tile w.
__global__ __launch_bounds__(256) void k_learn(
    const int* __restrict__ e_data, const int* __restrict__ at_data,
    const float* __restrict__ a_data,
    const float* __restrict__ e_tab, const float* __restrict__ at_tab,
    const _Float16* __restrict__ w1s, const float* __restrict__ b1,
    float* __restrict__ learn) {
  __shared__ __attribute__((aligned(16))) _Float16 ldsA[16 * 328];
  int row0 = blockIdx.x * 16;
  // 16 rows x 80 float4 groups (320 cols); all segments 16B aligned.
  for (int idx = threadIdx.x; idx < 16 * 80; idx += 256) {
    int m = idx / 80, c4 = idx - m * 80;
    int c = c4 << 2;
    int g = row0 + m;   // flat (b,s) row
    float4 v;
    if (c < 128)      v = ((const float4*)(e_tab + (size_t)e_data[g] * 128))[c4];
    else if (c < 256) v = ((const float4*)(at_tab + (size_t)at_data[g] * 128))[c4 - 32];
    else { float a = a_data[g]; v = make_float4(a, a, a, a); }
    store_h4(&ldsA[m * 328 + c], v);
  }
  __syncthreads();
  int w = threadIdx.x >> 5, lane = threadIdx.x & 31;
  v8f acc = {};
  for (int kt = 0; kt < 10; ++kt) {
    v16h a = load_a_lds(ldsA, 328, kt * 32);
    v16h b = load_b_swz(w1s, kt, w, 8);
    acc = wmma_f16(a, b, acc);
  }
  int col = (w << 4) + (lane & 15), hi = lane >> 4;
  float bb = b1[col];
  for (int r = 0; r < 8; ++r) {
    int g = row0 + r + 8 * hi;
    learn[g * 128 + col] = acc[r] + bb;
  }
}

// ---------------- K2 (per step): gates ----------------
// x = [learning_pre | it | learning | h_tilde_pre] (B x 512)
// LG = sigmoid(x@W3+b3) * (tanh(x@W2+b2)+1)/2
// gate_b = [LG | it] @ W4[128:384] + b4 ; also zero the h_tilde accumulator.
__global__ __launch_bounds__(256) void k_gates(
    const float* __restrict__ learn, const int* __restrict__ it_data,
    const float* __restrict__ it_tab,
    const float* __restrict__ ht_pre, float* __restrict__ ht_new,
    const _Float16* __restrict__ w2s, const float* __restrict__ b2,
    const _Float16* __restrict__ w3s, const float* __restrict__ b3,
    const _Float16* __restrict__ w4los, const float* __restrict__ b4,
    float* __restrict__ LGbuf, float* __restrict__ gatebuf, int t) {
  __shared__ __attribute__((aligned(16))) _Float16 ldsX[16 * 520];
  __shared__ __attribute__((aligned(16))) _Float16 ldsLI[16 * 264];
  int b0 = blockIdx.x * 16;
  // 16 rows x 128 float4 groups (512 cols)
  for (int idx = threadIdx.x; idx < 16 * 128; idx += 256) {
    int m = idx >> 7, c4 = idx & 127;
    int c = c4 << 2;
    int b = b0 + m;
    float4 v;
    if (c < 128) {
      v = (t > 0) ? ((const float4*)(learn + (size_t)(b * S_LEN + t - 1) * 128))[c4]
                  : make_float4(0.f, 0.f, 0.f, 0.f);
    } else if (c < 256) {
      v = ((const float4*)(it_tab + (size_t)it_data[b * S_LEN + t] * 128))[c4 - 32];
    } else if (c < 384) {
      v = ((const float4*)(learn + (size_t)(b * S_LEN + t) * 128))[c4 - 64];
    } else {
      v = ((const float4*)(ht_pre + (size_t)b * 128))[c4 - 96];
    }
    store_h4(&ldsX[m * 520 + c], v);
  }
  for (int idx = threadIdx.x; idx < 16 * 128; idx += 256)
    ht_new[b0 * 128 + idx] = 0.0f;
  __syncthreads();

  int w = threadIdx.x >> 5, lane = threadIdx.x & 31;
  v8f a2 = {}, a3 = {};
  for (int kt = 0; kt < 16; ++kt) {
    v16h a = load_a_lds(ldsX, 520, kt * 32);
    v16h bw2 = load_b_swz(w2s, kt, w, 8);
    a2 = wmma_f16(a, bw2, a2);
    v16h bw3 = load_b_swz(w3s, kt, w, 8);
    a3 = wmma_f16(a, bw3, a3);
  }
  int col = (w << 4) + (lane & 15), hi = lane >> 4;
  float bb2 = b2[col], bb3 = b3[col];
  for (int r = 0; r < 8; ++r) {
    int m = r + 8 * hi;
    int b = b0 + m;
    float lg  = tanhf(a2[r] + bb2);
    float gl  = sigf(a3[r] + bb3);
    float LGv = gl * (lg + 1.0f) * 0.5f;
    LGbuf[b * 128 + col] = LGv;
    ldsLI[m * 264 + col] = (_Float16)LGv;
    float itv = it_tab[it_data[b * S_LEN + t] * 128 + col];
    ldsLI[m * 264 + 128 + col] = (_Float16)itv;
  }
  __syncthreads();
  v8f a4 = {};
  for (int kt = 0; kt < 8; ++kt) {
    v16h a = load_a_lds(ldsLI, 264, kt * 32);
    v16h bw = load_b_swz(w4los, kt, w, 8);
    a4 = wmma_f16(a, bw, a4);
  }
  float bb4 = b4[col];
  for (int r = 0; r < 8; ++r) {
    int b = b0 + r + 8 * hi;
    gatebuf[b * 128 + col] = a4[r] + bb4;
  }
}

// ---------------- K3 (per step): memory update (the big GEMM) ----------------
// gamma_f = sigmoid(h_pre @ W4h + gate_b); h = q*LG + gamma_f*h_pre;
// h_tilde += q_next^T h (reduced via shfl + atomicAdd).
// grid = (33 n-tiles, 128 batch); 256 threads, wave w owns d-columns [16w,16w+16).
__global__ __launch_bounds__(256) void k_mem(
    float* __restrict__ h, const float* __restrict__ LGbuf,
    const float* __restrict__ gatebuf,
    const int* __restrict__ e_data, const float* __restrict__ q_matrix,
    const _Float16* __restrict__ w4hs, float* __restrict__ ht_new, int t) {
  __shared__ __attribute__((aligned(16))) float    hs[16 * 128];
  __shared__ __attribute__((aligned(16))) _Float16 hh[16 * 136];
  __shared__ float qv[16], qn[16];
  int b  = blockIdx.y;
  int n0 = blockIdx.x * 16;
  float* hrow = h + ((size_t)b * NQ_PAD + n0) * 128;
  // 512 float4 = 16 rows x 32 groups; b128 loads + b128/b64 LDS stores.
  for (int idx = threadIdx.x; idx < 512; idx += 256) {
    float4 v = ((const float4*)hrow)[idx];
    ((float4*)hs)[idx] = v;
    int m = idx >> 5, c4 = idx & 31;
    store_h4(&hh[m * 136 + (c4 << 2)], v);
  }
  if (threadIdx.x < 16) {
    int n = n0 + threadIdx.x;
    bool ok = n < NQ1;
    int ei  = e_data[b * S_LEN + t];
    int ei2 = e_data[b * S_LEN + t + 1];
    qv[threadIdx.x] = ok ? q_matrix[(size_t)ei  * NQ1 + n] : 0.0f;
    qn[threadIdx.x] = ok ? q_matrix[(size_t)ei2 * NQ1 + n] : 0.0f;
  }
  __syncthreads();

  int w = threadIdx.x >> 5, lane = threadIdx.x & 31;
  v8f acc = {};
  for (int kt = 0; kt < 4; ++kt) {
    v16h a  = load_a_lds(hh, 136, kt * 32);
    v16h bw = load_b_swz(w4hs, kt, w, 8);
    acc = wmma_f16(a, bw, acc);
  }
  int col = (w << 4) + (lane & 15), hi = lane >> 4;
  float gb  = gatebuf[b * 128 + col];
  float lgv = LGbuf[b * 128 + col];
  float psum = 0.0f;
  for (int r = 0; r < 8; ++r) {
    int m = r + 8 * hi;
    float gf   = sigf(acc[r] + gb);
    float hold = hs[m * 128 + col];
    float hnew = qv[m] * lgv + gf * hold;
    hrow[m * 128 + col] = hnew;
    psum += qn[m] * hnew;
  }
  psum += __shfl_xor(psum, 16, 32);
  if (hi == 0) atomicAdd(&ht_new[b * 128 + col], psum);
}

// ---------------- K4 (per step): prediction readout ----------------
// sum_j(([e_emb|h_tilde]@W5 + b5)[j]) == dot([e_emb|h_tilde], rowsum(W5)) + sum(b5)
__global__ void k_pred(const int* __restrict__ e_data, const float* __restrict__ e_tab,
                       const float* __restrict__ ht_new, const float* __restrict__ w5r,
                       float* __restrict__ out, int t) {
  int b = threadIdx.x;
  if (b < 128) {
    int ei = e_data[b * S_LEN + t];
    float dot = w5r[256];
    const float* er = e_tab + (size_t)ei * 128;
    for (int k = 0; k < 128; ++k) dot += er[k] * w5r[k];
    for (int k = 0; k < 128; ++k) dot += ht_new[b * 128 + k] * w5r[128 + k];
    out[b * S_LEN + t + 1] = sigf(dot * (1.0f / 128.0f));
  }
}

// ---------------- host-side launch ----------------

extern "C" void kernel_launch(void* const* d_in, const int* in_sizes, int n_in,
                              void* d_out, int out_size, void* d_ws, size_t ws_size,
                              hipStream_t stream) {
  (void)in_sizes; (void)n_in; (void)ws_size;
  const int*   e_data   = (const int*)d_in[0];
  const int*   at_data  = (const int*)d_in[1];
  const int*   it_data  = (const int*)d_in[2];
  const float* a_data   = (const float*)d_in[3];
  const float* q_matrix = (const float*)d_in[4];
  const float* at_tab   = (const float*)d_in[5];
  const float* it_tab   = (const float*)d_in[6];
  const float* e_tab    = (const float*)d_in[7];
  const float* W1 = (const float*)d_in[8];  const float* b1 = (const float*)d_in[9];
  const float* W2 = (const float*)d_in[10]; const float* b2 = (const float*)d_in[11];
  const float* W3 = (const float*)d_in[12]; const float* b3 = (const float*)d_in[13];
  const float* W4 = (const float*)d_in[14]; const float* b4 = (const float*)d_in[15];
  const float* W5 = (const float*)d_in[16]; const float* b5 = (const float*)d_in[17];
  float* out = (float*)d_out;

  char* ws = (char*)d_ws;
  size_t off = 0;
  auto allocF = [&](size_t n) { float* p = (float*)(ws + off); off += n * sizeof(float); return p; };
  float* h     = allocF((size_t)128 * NQ_PAD * 128);   // padded memory state
  float* ht0   = allocF(128 * 128);                    // h_tilde ping
  float* ht1   = allocF(128 * 128);                    // h_tilde pong
  float* learn = allocF((size_t)128 * S_LEN * 128);    // learning_all
  float* LGb   = allocF(128 * 128);
  float* gateb = allocF(128 * 128);
  float* w5r   = allocF(320);
  auto allocH = [&](size_t n) { _Float16* p = (_Float16*)(ws + off); off += n * sizeof(_Float16); return p; };
  _Float16* w1s   = allocH((size_t)10 * 8 * 512);
  _Float16* w2s   = allocH((size_t)16 * 8 * 512);
  _Float16* w3s   = allocH((size_t)16 * 8 * 512);
  _Float16* w4hs  = allocH((size_t)4 * 8 * 512);
  _Float16* w4los = allocH((size_t)8 * 8 * 512);

  // init: zero h + both h_tilde buffers (contiguous), zero output, prep weights
  long long nz = (long long)128 * NQ_PAD * 128 + 2LL * 128 * 128;
  k_zero<<<2048, 256, 0, stream>>>(h, nz);
  k_zero<<<32, 256, 0, stream>>>(out, (long long)out_size);
  k_w5row<<<2, 160, 0, stream>>>(W5, b5, w5r);
  k_swz<<<80, 256, 0, stream>>>(W1, w1s, 10);
  k_swz<<<128, 256, 0, stream>>>(W2, w2s, 16);
  k_swz<<<128, 256, 0, stream>>>(W3, w3s, 16);
  k_swz<<<32, 256, 0, stream>>>(W4, w4hs, 4);                 // W4h = rows [0,128)
  k_swz<<<64, 256, 0, stream>>>(W4 + 128 * 128, w4los, 8);    // W4l|W4i = rows [128,384)

  // hoisted GEMM
  k_learn<<<512, 256, 0, stream>>>(e_data, at_data, a_data, e_tab, at_tab, w1s, b1, learn);

  // sequential scan
  for (int t = 0; t < S_LEN - 1; ++t) {
    float* ht_pre = (t & 1) ? ht1 : ht0;
    float* ht_nxt = (t & 1) ? ht0 : ht1;
    k_gates<<<8, 256, 0, stream>>>(learn, it_data, it_tab, ht_pre, ht_nxt,
                                   w2s, b2, w3s, b3, w4los, b4, LGb, gateb, t);
    k_mem<<<dim3(33, 128), 256, 0, stream>>>(h, LGb, gateb, e_data, q_matrix,
                                             w4hs, ht_nxt, t);
    k_pred<<<1, 128, 0, stream>>>(e_data, e_tab, ht_nxt, w5r, out, t);
  }
}
